// GCN_55602646614506
// MI455X (gfx1250) — matrix operand order repaired and checked by reading, not compiled, not run on hardware
//
#include <hip/hip_runtime.h>

typedef __bf16 bf16;
typedef __attribute__((ext_vector_type(16))) __bf16 v16bf;
typedef __attribute__((ext_vector_type(8))) float v8f;

static __device__ __host__ constexpr int NN = 512;
static __device__ __host__ constexpr int NH = 32;
static __device__ __host__ constexpr int DD = 1024;
static __device__ __host__ constexpr int CC = 16;
static __device__ __host__ constexpr int PP = NH * NN;      // 16384 pairs
static __device__ __host__ constexpr int PKEEP = PP - NH;   // 16352
#define GAMA   0.9f
#define EPSI   1e-10f
#define LNEPS  1e-5f

// ---------------------------------------------------------------------------
// WMMA fragment loaders (layouts per CDNA5 ISA 7.12.2, 16-bit A 16x32 / B 32x16)
// ---------------------------------------------------------------------------
__device__ inline v16bf load_frag_a(const bf16* A, int lda, int row, int kb, int lane) {
    int grp = (lane >> 4) & 1;
    union { v16bf v; uint4 q[2]; } u;
    const bf16* base = A + (long)row * lda + kb + 8 * grp;
    u.q[0] = *(const uint4*)(base);        // halves 0..7  : K = kb + 8*grp + [0..8)
    u.q[1] = *(const uint4*)(base + 16);   // halves 8..15 : K = kb + 16 + 8*grp + [0..8)
    return u.v;
}
__device__ inline v16bf load_frag_b(const bf16* BT, int ldb, int col, int kb, int lane) {
    int grp = (lane >> 4) & 1;
    union { v16bf v; uint4 q[2]; } u;
    const bf16* base = BT + (long)col * ldb + kb + 8 * grp;  // K = kb + 8*grp + [0..8), kb + 16 + 8*grp + [0..8)
    u.q[0] = *(const uint4*)(base);
    u.q[1] = *(const uint4*)(base + 16);
    return u.v;
}

// ---------------------------------------------------------------------------
// bf16 WMMA GEMM, 64x32 tile per wave32 (4x2 register blocking):
//   C(M,N) = act(A(M,K) * BT(N,K)^T + bias), bf16 out
// 8 WMMAs per 6 fragment loads; ~112 live VGPRs -> no scratch spills.
// ---------------------------------------------------------------------------
__global__ void wmma_gemm_relu(const bf16* __restrict__ A, int lda,
                               const bf16* __restrict__ BT, int ldb,
                               const float* __restrict__ bias,
                               bf16* __restrict__ Cout, int ldc,
                               int M, int N, int K, int doRelu) {
    int wave = (blockIdx.x * blockDim.x + threadIdx.x) >> 5;
    int lane = threadIdx.x & 31;
    int nt32 = N >> 5;
    int mt = wave / nt32;
    int nt = wave - mt * nt32;
    if (mt >= (M >> 6)) return;  // wave-uniform guard (EXEC stays all-ones)
    int rbase = mt << 6;
    int cbase = nt << 5;
    int lrow = lane & 15;
    v8f acc[4][2];
    for (int i = 0; i < 4; ++i)
        for (int j = 0; j < 2; ++j)
            for (int r = 0; r < 8; ++r) acc[i][j][r] = 0.0f;
    for (int kb = 0; kb < K; kb += 32) {
        v16bf af[4], bfr[2];
#pragma unroll
        for (int i = 0; i < 4; ++i)
            af[i] = load_frag_a(A, lda, rbase + 16 * i + lrow, kb, lane);
#pragma unroll
        for (int j = 0; j < 2; ++j)
            bfr[j] = load_frag_b(BT, ldb, cbase + 16 * j + lrow, kb, lane);
#pragma unroll
        for (int i = 0; i < 4; ++i)
#pragma unroll
            for (int j = 0; j < 2; ++j)
                acc[i][j] = __builtin_amdgcn_wmma_f32_16x16x32_bf16(
                    false, af[i], false, bfr[j], (short)0, acc[i][j], false, false);
    }
    int grp = (lane >> 4) & 1;
    for (int j = 0; j < 2; ++j) {
        int col = cbase + 16 * j + lrow;
        float bv = bias ? bias[col] : 0.0f;
        for (int i = 0; i < 4; ++i) {
            for (int r = 0; r < 8; ++r) {
                int m = rbase + 16 * i + r + 8 * grp;
                float v = acc[i][j][r] + bv;
                if (doRelu) v = fmaxf(v, 0.0f);
                Cout[(long)m * ldc + col] = (bf16)v;
            }
        }
    }
}

// ---------------------------------------------------------------------------
// Fused dual WMMA GEMM for mbf hidden, 32x32 tile per wave (2x2 blocking, x2):
//   hdn[p, c*64+n] = relu( (a@w1_c + b1_c) * (sp@w2_c + b2_c) )
// ---------------------------------------------------------------------------
__global__ void wmma_dual_mbf(const bf16* __restrict__ Abuf,  // (P, 2048) bf16
                              const bf16* __restrict__ Sp,    // (P, 1024) bf16
                              const bf16* __restrict__ W1T,   // (C, 64, 2048) bf16
                              const bf16* __restrict__ W2T,   // (C, 64, 1024) bf16
                              const float* __restrict__ B1,   // (C, 64)
                              const float* __restrict__ B2,   // (C, 64)
                              bf16* __restrict__ Hdn) {       // (P, 1024) bf16
    int wave = (blockIdx.x * blockDim.x + threadIdx.x) >> 5;
    int lane = threadIdx.x & 31;
    int ns = wave & 1;           // which 32-col half of SUB=64
    int c  = (wave >> 1) & 15;
    int mt = wave >> 5;          // P/32 = 512
    int rbase = mt << 5;
    int cbase = ns << 5;
    int lrow = lane & 15;
    const bf16* w1 = W1T + (long)c * 64 * 2048;
    const bf16* w2 = W2T + (long)c * 64 * 1024;
    v8f acc1[2][2], acc2[2][2];
    for (int i = 0; i < 2; ++i)
        for (int j = 0; j < 2; ++j)
            for (int r = 0; r < 8; ++r) { acc1[i][j][r] = 0.0f; acc2[i][j][r] = 0.0f; }
    for (int kb = 0; kb < 2048; kb += 32) {
        v16bf af[2], bfr[2];
#pragma unroll
        for (int i = 0; i < 2; ++i)
            af[i] = load_frag_a(Abuf, 2048, rbase + 16 * i + lrow, kb, lane);
#pragma unroll
        for (int j = 0; j < 2; ++j)
            bfr[j] = load_frag_b(w1, 2048, cbase + 16 * j + lrow, kb, lane);
#pragma unroll
        for (int i = 0; i < 2; ++i)
#pragma unroll
            for (int j = 0; j < 2; ++j)
                acc1[i][j] = __builtin_amdgcn_wmma_f32_16x16x32_bf16(
                    false, af[i], false, bfr[j], (short)0, acc1[i][j], false, false);
    }
    for (int kb = 0; kb < 1024; kb += 32) {
        v16bf af[2], bfr[2];
#pragma unroll
        for (int i = 0; i < 2; ++i)
            af[i] = load_frag_a(Sp, 1024, rbase + 16 * i + lrow, kb, lane);
#pragma unroll
        for (int j = 0; j < 2; ++j)
            bfr[j] = load_frag_b(w2, 1024, cbase + 16 * j + lrow, kb, lane);
#pragma unroll
        for (int i = 0; i < 2; ++i)
#pragma unroll
            for (int j = 0; j < 2; ++j)
                acc2[i][j] = __builtin_amdgcn_wmma_f32_16x16x32_bf16(
                    false, af[i], false, bfr[j], (short)0, acc2[i][j], false, false);
    }
    int grp = (lane >> 4) & 1;
    for (int j = 0; j < 2; ++j) {
        int col = cbase + 16 * j + lrow;
        float bb1 = B1[c * 64 + col];
        float bb2 = B2[c * 64 + col];
        for (int i = 0; i < 2; ++i) {
            for (int r = 0; r < 8; ++r) {
                int m = rbase + 16 * i + r + 8 * grp;
                float v = fmaxf((acc1[i][j][r] + bb1) * (acc2[i][j][r] + bb2), 0.0f);
                Hdn[(long)m * 1024 + c * 64 + col] = (bf16)v;
            }
        }
    }
}

// ---------------------------------------------------------------------------
// Spatial pair features: (P, 36) padded to (P, 64) bf16
// ---------------------------------------------------------------------------
__global__ void feat_kernel(const float* __restrict__ coords,
                            const int* __restrict__ img_h, const int* __restrict__ img_w,
                            bf16* __restrict__ fp) {
    int p = blockIdx.x * blockDim.x + threadIdx.x;
    if (p >= PP) return;
    float h = (float)img_h[0], w = (float)img_w[0];
    int x = p >> 9, y = p & (NN - 1);
    const float* b1 = coords + x * 4;
    const float* b2 = coords + y * 4;
    float c1x = (b1[0] + b1[2]) * 0.5f, c1y = (b1[1] + b1[3]) * 0.5f;
    float c2x = (b2[0] + b2[2]) * 0.5f, c2y = (b2[1] + b2[3]) * 0.5f;
    float b1w = b1[2] - b1[0], b1h = b1[3] - b1[1];
    float b2w = b2[2] - b2[0], b2h = b2[3] - b2[1];
    float dx = fabsf(c2x - c1x) / (b1w + EPSI);
    float dy = fabsf(c2y - c1y) / (b1h + EPSI);
    float ltx = fmaxf(b1[0], b2[0]), lty = fmaxf(b1[1], b2[1]);
    float rbx = fminf(b1[2], b2[2]), rby = fminf(b1[3], b2[3]);
    float inter = fmaxf(rbx - ltx, 0.0f) * fmaxf(rby - lty, 0.0f);
    float a1 = b1w * b1h, a2 = b2w * b2h;
    float iou = inter / (a1 + a2 - inter);
    float f[18] = { c1x / w, c1y / h, c2x / w, c2y / h,
                    b1w / w, b1h / h, b2w / w, b2h / h,
                    a1 / (h * w), a2 / (h * w), a2 / (a1 + EPSI),
                    b1w / (b1h + EPSI), b2w / (b2h + EPSI), iou,
                    (c2x > c1x) ? dx : 0.0f, (c2x < c1x) ? dx : 0.0f,
                    (c2y > c1y) ? dy : 0.0f, (c2y < c1y) ? dy : 0.0f };
    bf16* row = fp + (long)p * 64;
    for (int j = 0; j < 18; ++j) {
        row[j] = (bf16)f[j];
        row[18 + j] = (bf16)logf(f[j] + EPSI);
    }
    for (int j = 36; j < 64; ++j) row[j] = (bf16)0.0f;
}

// ---------------------------------------------------------------------------
// Weight prep kernels
// ---------------------------------------------------------------------------
// src (Kr, Nc) f32 row-major -> dst (Nc, Kpad) bf16 (transposed, K zero-padded)
__global__ void transpose_to_bf16(const float* __restrict__ src, bf16* __restrict__ dst,
                                  int Kr, int Nc, int Kpad) {
    long idx = (long)blockIdx.x * blockDim.x + threadIdx.x;
    long total = (long)Nc * Kpad;
    if (idx >= total) return;
    int o = (int)(idx / Kpad), k = (int)(idx % Kpad);
    float v = (k < Kr) ? src[(long)k * Nc + o] : 0.0f;
    dst[idx] = (bf16)v;
}
// src (Cb, K, Nc) -> dst (Cb, Nc, K)
__global__ void transpose_batched(const float* __restrict__ src, bf16* __restrict__ dst,
                                  int Cb, int K, int Nc) {
    long idx = (long)blockIdx.x * blockDim.x + threadIdx.x;
    long total = (long)Cb * K * Nc;
    if (idx >= total) return;
    int c = (int)(idx / ((long)Nc * K));
    long rem = idx - (long)c * Nc * K;
    int o = (int)(rem / K), k = (int)(rem % K);
    dst[idx] = (bf16)src[((long)c * K + k) * Nc + o];
}
__global__ void b3sum_kernel(const float* __restrict__ b3, float* __restrict__ out) {
    int o = blockIdx.x * blockDim.x + threadIdx.x;
    if (o >= 1024) return;
    float s = 0.0f;
    for (int c = 0; c < CC; ++c) s += b3[c * 1024 + o];
    out[o] = s;
}
__global__ void copy_f32(const float* __restrict__ src, float* __restrict__ dst, long n) {
    long i = (long)blockIdx.x * blockDim.x + threadIdx.x;
    if (i < n) dst[i] = src[i];
}

// a[p, 0:1024] = human[x], a[p, 1024:2048] = node[y]  (bf16), 8 elems/thread
__global__ void gather_a_kernel(const float* __restrict__ human, const float* __restrict__ node,
                                bf16* __restrict__ a) {
    long idx = (long)blockIdx.x * blockDim.x + threadIdx.x;  // PP*256 threads
    if (idx >= (long)PP * 256) return;
    int p = (int)(idx >> 8);
    int d8 = ((int)idx & 255) << 3;
    int x = p >> 9, y = p & (NN - 1);
    const float* src = (d8 < 1024) ? (human + (long)x * 1024 + d8)
                                   : (node + (long)y * 1024 + (d8 - 1024));
    float4 v0 = *(const float4*)(src);
    float4 v1 = *(const float4*)(src + 4);
    union { uint4 q; bf16 e[8]; } u;
    u.e[0] = (bf16)v0.x; u.e[1] = (bf16)v0.y; u.e[2] = (bf16)v0.z; u.e[3] = (bf16)v0.w;
    u.e[4] = (bf16)v1.x; u.e[5] = (bf16)v1.y; u.e[6] = (bf16)v1.z; u.e[7] = (bf16)v1.w;
    *(uint4*)(a + (long)p * 2048 + d8) = u.q;
}

// adjacency[p] = relu(sum_o weight[p,o]*adj_w[o] + adj_b)   (vectorized 8 bf16/load)
__global__ void adj_kernel(const bf16* __restrict__ Wgt, const float* __restrict__ adj_w,
                           const float* __restrict__ adj_b, float* __restrict__ Adj) {
    int p = blockIdx.x * blockDim.x + threadIdx.x;
    if (p >= PP) return;
    const uint4* row = (const uint4*)(Wgt + (long)p * 1024);
    float s = 0.0f;
    for (int o8 = 0; o8 < 128; ++o8) {
        union { uint4 q; bf16 e[8]; } u;
        u.q = row[o8];
        const float* wv = adj_w + o8 * 8;
        for (int j = 0; j < 8; ++j) s += (float)u.e[j] * wv[j];
    }
    Adj[p] = fmaxf(s + adj_b[0], 0.0f);
}

// hn_msg (NH, D) = Adj(NH,N) @ node(N,D)
__global__ void hn_msg_kernel(const float* __restrict__ Adj, const float* __restrict__ node,
                              float* __restrict__ msg) {
    int idx = blockIdx.x * blockDim.x + threadIdx.x;
    if (idx >= NH * DD) return;
    int i = idx >> 10, d = idx & 1023;
    float s = 0.0f;
    for (int y = 0; y < NN; ++y) s += Adj[i * NN + y] * node[(long)y * DD + d];
    msg[idx] = s;
}
// nd_msg (N, D) = Adj^T @ human(NH,D)
__global__ void nd_msg_kernel(const float* __restrict__ Adj, const float* __restrict__ human,
                              float* __restrict__ msg) {
    int idx = blockIdx.x * blockDim.x + threadIdx.x;
    if (idx >= NN * DD) return;
    int y = idx >> 10, d = idx & 1023;
    float s = 0.0f;
    for (int i = 0; i < NH; ++i) s += Adj[i * NN + y] * human[(long)i * DD + d];
    msg[idx] = s;
}

// out_row = LN(old*GAMA + (1-GAMA)*msg) * g + b ; one block (256 thr) per row, D=1024
__global__ void update_ln(const float* __restrict__ oldv, const float* __restrict__ msg,
                          const float* __restrict__ g, const float* __restrict__ b,
                          float* __restrict__ outv) {
    int row = blockIdx.x, t = threadIdx.x;
    __shared__ float red[256];
    float loc[4]; float s = 0.0f;
    for (int j = 0; j < 4; ++j) {
        int d = t + j * 256;
        float v = oldv[(long)row * DD + d] * GAMA + (1.0f - GAMA) * msg[(long)row * DD + d];
        loc[j] = v; s += v;
    }
    red[t] = s; __syncthreads();
    for (int off = 128; off > 0; off >>= 1) { if (t < off) red[t] += red[t + off]; __syncthreads(); }
    float mean = red[0] * (1.0f / DD); __syncthreads();
    float vs = 0.0f;
    for (int j = 0; j < 4; ++j) { float dv = loc[j] - mean; vs += dv * dv; }
    red[t] = vs; __syncthreads();
    for (int off = 128; off > 0; off >>= 1) { if (t < off) red[t] += red[t + off]; __syncthreads(); }
    float inv = rsqrtf(red[0] * (1.0f / DD) + LNEPS);
    for (int j = 0; j < 4; ++j) {
        int d = t + j * 256;
        outv[(long)row * DD + d] = (loc[j] - mean) * inv * g[d] + b[d];
    }
}

// scatter kept pairs: k -> (x = k/511, y = r + (r>=x)), write 5 outputs
__global__ void final_out(const float* __restrict__ human, const float* __restrict__ node,
                          const float* __restrict__ Adj, float* __restrict__ out) {
    int k = blockIdx.x;
    int x = k / (NN - 1);
    int r = k - x * (NN - 1);
    int y = r + (r >= x ? 1 : 0);
    const long O1 = (long)PKEEP * DD;
    for (int d = threadIdx.x; d < DD; d += blockDim.x) {
        out[(long)k * DD + d]      = human[(long)x * DD + d];
        out[O1 + (long)k * DD + d] = node[(long)y * DD + d];
    }
    if (threadIdx.x == 0) {
        out[2 * O1 + k]             = Adj[x * NN + y];
        out[2 * O1 + PKEEP + k]     = (float)x;
        out[2 * O1 + 2 * PKEEP + k] = (float)y;
    }
}

// ---------------------------------------------------------------------------
extern "C" void kernel_launch(void* const* d_in, const int* in_sizes, int n_in,
                              void* d_out, int out_size, void* d_ws, size_t ws_size,
                              hipStream_t stream) {
    const float* node_in = (const float*)d_in[1];
    const float* coords  = (const float*)d_in[2];
    const int*   img_h   = (const int*)d_in[3];
    const int*   img_w   = (const int*)d_in[4];
    const float* sp_w1   = (const float*)d_in[5];
    const float* sp_b1   = (const float*)d_in[6];
    const float* sp_w2   = (const float*)d_in[7];
    const float* sp_b2   = (const float*)d_in[8];
    const float* sp_w3   = (const float*)d_in[9];
    const float* sp_b3   = (const float*)d_in[10];
    const float* att_w1  = (const float*)d_in[11];
    const float* att_b1  = (const float*)d_in[12];
    const float* att_w2  = (const float*)d_in[13];
    const float* att_b2  = (const float*)d_in[14];
    const float* att_w3  = (const float*)d_in[15];
    const float* att_b3  = (const float*)d_in[16];
    const float* adj_w   = (const float*)d_in[17];
    const float* adj_b   = (const float*)d_in[18];
    const float* lnh_g   = (const float*)d_in[19];
    const float* lnh_b   = (const float*)d_in[20];
    const float* lno_g   = (const float*)d_in[21];
    const float* lno_b   = (const float*)d_in[22];

    char* ws = (char*)d_ws;
    size_t off = 0;
    auto alloc = [&](size_t bytes) -> void* {
        void* p = ws + off;
        off = (off + bytes + 255) & ~(size_t)255;
        return p;
    };
    bf16* fpad  = (bf16*)alloc((size_t)PP * 64 * 2);
    bf16* sp1   = (bf16*)alloc((size_t)PP * 128 * 2);
    bf16* sp2   = (bf16*)alloc((size_t)PP * 256 * 2);
    bf16* spv   = (bf16*)alloc((size_t)PP * 1024 * 2);
    bf16* a_bf  = (bf16*)alloc((size_t)PP * 2048 * 2);
    bf16* hdn   = (bf16*)alloc((size_t)PP * 1024 * 2);
    bf16* wgt   = (bf16*)alloc((size_t)PP * 1024 * 2);
    float* Adj  = (float*)alloc((size_t)PP * 4);
    bf16* spw1T = (bf16*)alloc((size_t)128 * 64 * 2);
    bf16* spw2T = (bf16*)alloc((size_t)256 * 128 * 2);
    bf16* spw3T = (bf16*)alloc((size_t)1024 * 256 * 2);
    bf16* w1T   = (bf16*)alloc((size_t)CC * 64 * 2048 * 2);
    bf16* w2T   = (bf16*)alloc((size_t)CC * 64 * 1024 * 2);
    bf16* w3T   = (bf16*)alloc((size_t)1024 * 1024 * 2);
    float* b3s  = (float*)alloc(1024 * 4);
    float* nodeA  = (float*)alloc((size_t)NN * DD * 4);
    float* nodeB  = (float*)alloc((size_t)NN * DD * 4);
    float* nodeC  = (float*)alloc((size_t)NN * DD * 4);
    float* humA   = (float*)alloc((size_t)NH * DD * 4);
    float* humB   = (float*)alloc((size_t)NH * DD * 4);
    float* humC   = (float*)alloc((size_t)NH * DD * 4);
    float* hnMsg  = (float*)alloc((size_t)NH * DD * 4);
    float* ndMsg  = (float*)alloc((size_t)NN * DD * 4);

    const int TB = 256;
    auto blocks = [](long n, int tb) { return (int)((n + tb - 1) / tb); };
    // 64x32-tile GEMM grid: one wave per tile, 8 waves per block
    auto gemm_blocks = [](int M, int N) { return ((M >> 6) * (N >> 5)) / 8; };

    // ---- prep: features + weight conversion ----
    feat_kernel<<<blocks(PP, TB), TB, 0, stream>>>(coords, img_h, img_w, fpad);
    transpose_to_bf16<<<blocks((long)128 * 64, TB), TB, 0, stream>>>(sp_w1, spw1T, 36, 128, 64);
    transpose_to_bf16<<<blocks((long)256 * 128, TB), TB, 0, stream>>>(sp_w2, spw2T, 128, 256, 128);
    transpose_to_bf16<<<blocks((long)1024 * 256, TB), TB, 0, stream>>>(sp_w3, spw3T, 256, 1024, 256);
    transpose_to_bf16<<<blocks((long)1024 * 1024, TB), TB, 0, stream>>>(att_w3, w3T, 1024, 1024, 1024);
    transpose_batched<<<blocks((long)CC * 2048 * 64, TB), TB, 0, stream>>>(att_w1, w1T, CC, 2048, 64);
    transpose_batched<<<blocks((long)CC * 1024 * 64, TB), TB, 0, stream>>>(att_w2, w2T, CC, 1024, 64);
    b3sum_kernel<<<4, TB, 0, stream>>>(att_b3, b3s);
    copy_f32<<<blocks((long)NN * DD, TB), TB, 0, stream>>>(node_in, nodeA, (long)NN * DD);
    copy_f32<<<blocks((long)NH * DD, TB), TB, 0, stream>>>(node_in, humA, (long)NH * DD);

    // ---- spatial MLP: sp = relu(relu(relu(f@W1+b1)@W2+b2)@W3+b3), all WMMA ----
    wmma_gemm_relu<<<gemm_blocks(PP, 128), TB, 0, stream>>>(
        fpad, 64, spw1T, 64, sp_b1, sp1, 128, PP, 128, 64, 1);
    wmma_gemm_relu<<<gemm_blocks(PP, 256), TB, 0, stream>>>(
        sp1, 128, spw2T, 128, sp_b2, sp2, 256, PP, 256, 128, 1);
    wmma_gemm_relu<<<gemm_blocks(PP, 1024), TB, 0, stream>>>(
        sp2, 256, spw3T, 256, sp_b3, spv, 1024, PP, 1024, 256, 1);

    // ---- 2 message-passing iterations ----
    auto iterate = [&](const float* hIn, const float* nIn, float* hOut, float* nOut) {
        gather_a_kernel<<<blocks((long)PP * 256, TB), TB, 0, stream>>>(hIn, nIn, a_bf);
        wmma_dual_mbf<<<(PP / 32) * CC * 2 / 8, TB, 0, stream>>>(
            a_bf, spv, w1T, w2T, att_b1, att_b2, hdn);
        wmma_gemm_relu<<<gemm_blocks(PP, 1024), TB, 0, stream>>>(
            hdn, 1024, w3T, 1024, b3s, wgt, 1024, PP, 1024, 1024, 1);
        adj_kernel<<<blocks(PP, TB), TB, 0, stream>>>(wgt, adj_w, adj_b, Adj);
        hn_msg_kernel<<<blocks((long)NH * DD, TB), TB, 0, stream>>>(Adj, nIn, hnMsg);
        nd_msg_kernel<<<blocks((long)NN * DD, TB), TB, 0, stream>>>(Adj, hIn, ndMsg);
        update_ln<<<NN, TB, 0, stream>>>(nIn, ndMsg, lno_g, lno_b, nOut);
        update_ln<<<NH, TB, 0, stream>>>(hIn, hnMsg, lnh_g, lnh_b, hOut);
    };
    iterate(humA, nodeA, humB, nodeB);
    iterate(humB, nodeB, humC, nodeC);

    // ---- gather kept pairs into d_out ----
    final_out<<<PKEEP, TB, 0, stream>>>(humC, nodeC, Adj, (float*)d_out);
    (void)in_sizes; (void)n_in; (void)out_size; (void)ws_size;
}
